// GPTLanguageModel_75368086110868
// MI455X (gfx1250) — compile-verified, matrix operand order checked by reading, not run on hardware
//
#include <hip/hip_runtime.h>
#include <hip/hip_bf16.h>

#define BB   2
#define TT   2048
#define VV   32000
#define DD   768
#define NH   12
#define NKV  4
#define HD   64
#define FFN  2048
#define LL   6
#define MM   (BB * TT)

typedef __attribute__((ext_vector_type(8)))  int      v8i;
typedef __attribute__((ext_vector_type(8)))  float    v8f;
typedef __attribute__((ext_vector_type(16))) _Float16 v16h;

#if defined(__AMDGCN__)
#  if defined(__has_builtin)
#    if __has_builtin(__builtin_amdgcn_tensor_load_to_lds)
#      define HAVE_TDM 1
#    endif
#  endif
#endif
#ifndef HAVE_TDM
#  define HAVE_TDM 0
#endif

#if HAVE_TDM
typedef __attribute__((ext_vector_type(4))) unsigned int tdm_g0_t;
typedef __attribute__((ext_vector_type(8))) int          tdm_g1_t;
typedef __attribute__((ext_vector_type(4))) int          tdm_g23_t;

__device__ inline void tdm_load_tile_2d(unsigned ldsOff, const void* gptr,
                                        unsigned rowBytes, unsigned rows,
                                        unsigned tileW, unsigned tileH,
                                        unsigned long long strideBytes) {
  unsigned long long ga = (unsigned long long)(uintptr_t)gptr;
  tdm_g0_t g0;
  g0[0] = 1u;
  g0[1] = ldsOff;
  g0[2] = (unsigned)ga;
  g0[3] = (unsigned)(ga >> 32) | (2u << 30);
  tdm_g1_t g1;
  g1[0] = 0;
  g1[1] = (int)((rowBytes & 0xFFFFu) << 16);
  g1[2] = (int)((rowBytes >> 16) | ((rows & 0xFFFFu) << 16));
  g1[3] = (int)((rows >> 16) | (tileW << 16));
  g1[4] = (int)tileH;
  g1[5] = (int)(unsigned)(strideBytes & 0xFFFFFFFFull);
  g1[6] = (int)(unsigned)((strideBytes >> 32) & 0xFFFFull);
  g1[7] = 0;
  tdm_g23_t z4 = {0, 0, 0, 0};
#if __clang_major__ >= 23
  tdm_g1_t z8 = {0, 0, 0, 0, 0, 0, 0, 0};
  __builtin_amdgcn_tensor_load_to_lds(g0, g1, z4, z4, z8, 0);
#else
  __builtin_amdgcn_tensor_load_to_lds(g0, g1, z4, z4, 0);
#endif
}
#endif


__device__ inline v8i load_frag_i8(const signed char* base, int lane) {
  const long long* rp = (const long long*)(base + (size_t)(lane & 15) * 64 +
                                           ((lane >> 4) << 3));
  long long c0 = rp[0], c1 = rp[2], c2 = rp[4], c3 = rp[6];
  v8i r;
  r[0] = (int)c0; r[1] = (int)(c0 >> 32);
  r[2] = (int)c1; r[3] = (int)(c1 >> 32);
  r[4] = (int)c2; r[5] = (int)(c2 >> 32);
  r[6] = (int)c3; r[7] = (int)(c3 >> 32);
  return r;
}

__device__ inline v16h load_frag_f16(const _Float16* base, int strideH,
                                     int lane, int koff) {
  const _Float16* rp = base + (size_t)(lane & 15) * strideH + koff +
                       ((lane >> 4) << 3);
  v16h r;
#pragma unroll
  for (int g = 0; g < 2; ++g)
#pragma unroll
    for (int e = 0; e < 8; ++e) r[g * 8 + e] = rp[g * 16 + e];
  return r;
}


__global__ void __launch_bounds__(256) embed_kernel(
    const int* __restrict__ idx, const float* __restrict__ emb,
    const float* __restrict__ pos, float* __restrict__ x) {
  const int m = blockIdx.x;
  const int t = m % TT;
  const int tok = idx[m];
  const float* er = emb + (size_t)tok * DD;
  const float* pr = pos + (size_t)t * DD;
  float* xr = x + (size_t)m * DD;
  for (int i = threadIdx.x; i < DD; i += 256) xr[i] = er[i] + pr[i];
}

__global__ void __launch_bounds__(256) rmsnorm_kernel(
    const float* __restrict__ x, const float* __restrict__ w,
    float* __restrict__ y, int K) {
  const int row = blockIdx.x;
  const float* xr = x + (size_t)row * K;
  __shared__ float red[256];
  float s = 0.0f;
  for (int i = threadIdx.x; i < K; i += 256) { float v = xr[i]; s += v * v; }
  red[threadIdx.x] = s;
  __syncthreads();
  for (int o = 128; o > 0; o >>= 1) {
    if (threadIdx.x < o) red[threadIdx.x] += red[threadIdx.x + o];
    __syncthreads();
  }
  const float inv = rsqrtf(red[0] / (float)K);
  float* yr = y + (size_t)row * K;
  for (int i = threadIdx.x; i < K; i += 256) yr[i] = w[i] * xr[i] * inv;
}

__global__ void __launch_bounds__(256) act_quant_kernel(
    const float* __restrict__ x, signed char* __restrict__ q,
    float* __restrict__ sc, int K) {
  const int row = blockIdx.x;
  const float* xr = x + (size_t)row * K;
  __shared__ float red[256];
  float mx = 0.0f;
  for (int i = threadIdx.x; i < K; i += 256) mx = fmaxf(mx, fabsf(xr[i]));
  red[threadIdx.x] = mx;
  __syncthreads();
  for (int o = 128; o > 0; o >>= 1) {
    if (threadIdx.x < o)
      red[threadIdx.x] = fmaxf(red[threadIdx.x], red[threadIdx.x + o]);
    __syncthreads();
  }
  const float amax = fmaxf(red[0], 1e-5f);
  const float s = 127.0f / amax;
  signed char* qr = q + (size_t)row * K;
  for (int i = threadIdx.x; i < K; i += 256) {
    float v = rintf(xr[i] * s);
    v = fminf(fmaxf(v, -128.0f), 127.0f);
    qr[i] = (signed char)v;
  }
  if (threadIdx.x == 0) sc[row] = amax / 127.0f;
}

__global__ void __launch_bounds__(256) wabs_reduce_kernel(
    const float* __restrict__ w, long long n, float* __restrict__ partial) {
  __shared__ float red[256];
  float s = 0.0f;
  for (long long i = (long long)blockIdx.x * 256 + threadIdx.x; i < n;
       i += (long long)gridDim.x * 256)
    s += fabsf(w[i]);
  red[threadIdx.x] = s;
  __syncthreads();
  for (int o = 128; o > 0; o >>= 1) {
    if (threadIdx.x < o) red[threadIdx.x] += red[threadIdx.x + o];
    __syncthreads();
  }
  if (threadIdx.x == 0) partial[blockIdx.x] = red[0];
}

__global__ void __launch_bounds__(256) wquant_kernel(
    const float* __restrict__ w, long long n,
    const float* __restrict__ partial, signed char* __restrict__ q,
    float* __restrict__ wscale) {
  __shared__ float red[256];
  red[threadIdx.x] = partial[threadIdx.x];
  __syncthreads();
  for (int o = 128; o > 0; o >>= 1) {
    if (threadIdx.x < o) red[threadIdx.x] += red[threadIdx.x + o];
    __syncthreads();
  }
  const float mean = fmaxf(red[0] / (float)n, 1e-5f);
  const float inv = 1.0f / mean;
  for (long long i = (long long)blockIdx.x * 256 + threadIdx.x; i < n;
       i += (long long)gridDim.x * 256) {
    float v = rintf(w[i] * inv);
    v = fminf(fmaxf(v, -1.0f), 1.0f);
    q[i] = (signed char)v;
  }
  if (blockIdx.x == 0 && threadIdx.x == 0) *wscale = mean;
}

__global__ void __launch_bounds__(256) silu_mul_kernel(
    float* __restrict__ g, const float* __restrict__ u, long long n) {
  for (long long i = (long long)blockIdx.x * 256 + threadIdx.x; i < n;
       i += (long long)gridDim.x * 256) {
    float gv = g[i];
    g[i] = gv * (1.0f / (1.0f + expf(-gv))) * u[i];
  }
}

__global__ void zero_loss_kernel(float* p) {
  if (threadIdx.x == 0 && blockIdx.x == 0) *p = 0.0f;
}

__global__ void __launch_bounds__(256) loss_kernel(
    const float* __restrict__ logits, const int* __restrict__ tgt,
    float* lossAcc, float invM) {
  const int row = blockIdx.x;
  const float* lr = logits + (size_t)row * VV;
  __shared__ float red[256];
  float mx = -3.0e38f;
  for (int i = threadIdx.x; i < VV; i += 256) mx = fmaxf(mx, lr[i]);
  red[threadIdx.x] = mx;
  __syncthreads();
  for (int o = 128; o > 0; o >>= 1) {
    if (threadIdx.x < o)
      red[threadIdx.x] = fmaxf(red[threadIdx.x], red[threadIdx.x + o]);
    __syncthreads();
  }
  mx = red[0];
  __syncthreads();
  float s = 0.0f;
  for (int i = threadIdx.x; i < VV; i += 256) s += expf(lr[i] - mx);
  red[threadIdx.x] = s;
  __syncthreads();
  for (int o = 128; o > 0; o >>= 1) {
    if (threadIdx.x < o) red[threadIdx.x] += red[threadIdx.x + o];
    __syncthreads();
  }
  if (threadIdx.x == 0) {
    float lp = lr[tgt[row]] - mx - logf(red[0]);
    atomicAdd(lossAcc, -lp * invM);
  }
}

__global__ void __launch_bounds__(256) gemm_i8_kernel(
    const signed char* __restrict__ Aq, const float* __restrict__ asc,
    const signed char* __restrict__ Bq, const float* __restrict__ wscp,
    const float* res, float* C, int M, int N, int K, int useRes) {
  __shared__ signed char lA[64][64];
  __shared__ signed char lB[64][64];
  const int m0 = blockIdx.y * 64;
  const int n0 = blockIdx.x * 64;
  const int tid = threadIdx.x;
  const int lane = tid & 31;
  const int w = tid >> 5;
  const int wm = (w >> 1) << 4;
  const int wn = (w & 1) << 5;

  v8i acc0 = {0, 0, 0, 0, 0, 0, 0, 0};
  v8i acc1 = {0, 0, 0, 0, 0, 0, 0, 0};

#if HAVE_TDM
  const unsigned ldsA = (unsigned)(uintptr_t)&lA[0][0];
  const unsigned ldsB = (unsigned)(uintptr_t)&lB[0][0];
#endif

  for (int kt = 0; kt < K; kt += 64) {
    __syncthreads();
#if HAVE_TDM
    if (tid < 32) {
      tdm_load_tile_2d(ldsA, Aq + (size_t)m0 * K + kt, (unsigned)K,
                       (unsigned)M, 64, 64, (unsigned long long)K);
      tdm_load_tile_2d(ldsB, Bq + (size_t)n0 * K + kt, (unsigned)K,
                       (unsigned)N, 64, 64, (unsigned long long)K);
      __builtin_amdgcn_s_wait_tensorcnt(0);
    }
#else
    {
      const int lr = tid >> 2, lc = (tid & 3) << 4;
      *(int4*)&lA[lr][lc] = *(const int4*)(Aq + (size_t)(m0 + lr) * K + kt + lc);
      *(int4*)&lB[lr][lc] = *(const int4*)(Bq + (size_t)(n0 + lr) * K + kt + lc);
      if (kt + 64 < K) {
        __builtin_prefetch(Aq + (size_t)(m0 + lr) * K + kt + 64 + lc, 0, 1);
        __builtin_prefetch(Bq + (size_t)(n0 + lr) * K + kt + 64 + lc, 0, 1);
      }
    }
#endif
    __syncthreads();
    v8i a  = load_frag_i8(&lA[wm][0], lane);
    v8i b0 = load_frag_i8(&lB[wn][0], lane);
    v8i b1 = load_frag_i8(&lB[wn + 16][0], lane);
    acc0 = __builtin_amdgcn_wmma_i32_16x16x64_iu8(true, a, true, b0, acc0,
                                                  false, false);
    acc1 = __builtin_amdgcn_wmma_i32_16x16x64_iu8(true, a, true, b1, acc1,
                                                  false, false);
  }

  const float wscale = wscp[0];
  const int rowLo = m0 + wm + ((lane >> 4) << 3);
  const int col0 = n0 + wn + (lane & 15);
#pragma unroll
  for (int r = 0; r < 8; ++r) {
    const int row = rowLo + r;
    const float sa = asc[row] * wscale;
    float y0 = (float)acc0[r] * sa;
    float y1 = (float)acc1[r] * sa;
    if (useRes) {
      y0 += res[(size_t)row * N + col0];
      y1 += res[(size_t)row * N + col0 + 16];
    }
    C[(size_t)row * N + col0] = y0;
    C[(size_t)row * N + col0 + 16] = y1;
  }
}

__global__ void __launch_bounds__(32) flash_attn_kernel(
    const float* __restrict__ q, const float* __restrict__ k,
    const float* __restrict__ v, float* __restrict__ o) {
  const int lane = threadIdx.x;
  const int qt = blockIdx.x, h = blockIdx.y, b = blockIdx.z;
  const int hkv = h / (NH / NKV);
  const int qbase = qt * 16;

  __shared__ _Float16 sQ[16][64];
  __shared__ _Float16 sK[32][64];
  __shared__ _Float16 sVt[64][32];
  __shared__ _Float16 sP[16][32];

  {
    const int r = lane >> 1, c0 = (lane & 1) * 32;
    const float* qp = q + ((size_t)(b * TT + qbase + r)) * (NH * HD) + h * HD + c0;
#pragma unroll
    for (int i = 0; i < 32; ++i) sQ[r][c0 + i] = (_Float16)(qp[i] * 0.125f);
  }
  __syncthreads();
  const v16h aQ0 = load_frag_f16(&sQ[0][0], 64, lane, 0);
  const v16h aQ1 = load_frag_f16(&sQ[0][0], 64, lane, 32);

  float mprev[8], lsum[8];
#pragma unroll
  for (int r = 0; r < 8; ++r) { mprev[r] = -3.0e38f; lsum[r] = 0.0f; }
  v8f O0 = {}, O1 = {}, O2 = {}, O3 = {};

  const int hh = lane >> 4, nn = lane & 15;
  const int nkt = (qbase + 16 + 31) / 32;

  for (int kt = 0; kt < nkt; ++kt) {
    __syncthreads();
    {
      const int t = kt * 32 + lane;
      const float* kp = k + ((size_t)(b * TT + t)) * (NKV * HD) + hkv * HD;
      const float* vp = v + ((size_t)(b * TT + t)) * (NKV * HD) + hkv * HD;
#pragma unroll
      for (int d = 0; d < 64; ++d) {
        sK[lane][d] = (_Float16)kp[d];
        sVt[d][lane] = (_Float16)vp[d];
      }
    }
    __syncthreads();

    v8f S0 = {}, S1 = {};
    {
      v16h b0a = load_frag_f16(&sK[0][0], 64, lane, 0);
      v16h b0b = load_frag_f16(&sK[0][0], 64, lane, 32);
      S0 = __builtin_amdgcn_wmma_f32_16x16x32_f16(false, aQ0, false, b0a,
                                                  (short)0, S0, false, false);
      S0 = __builtin_amdgcn_wmma_f32_16x16x32_f16(false, aQ1, false, b0b,
                                                  (short)0, S0, false, false);
      v16h b1a = load_frag_f16(&sK[16][0], 64, lane, 0);
      v16h b1b = load_frag_f16(&sK[16][0], 64, lane, 32);
      S1 = __builtin_amdgcn_wmma_f32_16x16x32_f16(false, aQ0, false, b1a,
                                                  (short)0, S1, false, false);
      S1 = __builtin_amdgcn_wmma_f32_16x16x32_f16(false, aQ1, false, b1b,
                                                  (short)0, S1, false, false);
    }

    const int kc0 = kt * 32 + nn, kc1 = kt * 32 + 16 + nn;
#pragma unroll
    for (int r = 0; r < 8; ++r) {
      const int qrow = qbase + r + hh * 8;
      float s0 = (kc0 <= qrow) ? S0[r] : -3.0e38f;
      float s1 = (kc1 <= qrow) ? S1[r] : -3.0e38f;
      float mloc = fmaxf(s0, s1);
#pragma unroll
      for (int off = 8; off > 0; off >>= 1)
        mloc = fmaxf(mloc, __shfl_xor(mloc, off, 16));
      const float mnew = fmaxf(mprev[r], mloc);
      const float alpha = expf(mprev[r] - mnew);
      const float p0 = expf(s0 - mnew), p1 = expf(s1 - mnew);
      float ps = p0 + p1;
#pragma unroll
      for (int off = 8; off > 0; off >>= 1) ps += __shfl_xor(ps, off, 16);
      lsum[r] = lsum[r] * alpha + ps;
      mprev[r] = mnew;
      O0[r] *= alpha; O1[r] *= alpha; O2[r] *= alpha; O3[r] *= alpha;
      sP[r + hh * 8][nn] = (_Float16)p0;
      sP[r + hh * 8][16 + nn] = (_Float16)p1;
    }
    __syncthreads();

    const v16h aP = load_frag_f16(&sP[0][0], 32, lane, 0);
    v16h bV0 = load_frag_f16(&sVt[0][0], 32, lane, 0);
    v16h bV1 = load_frag_f16(&sVt[16][0], 32, lane, 0);
    v16h bV2 = load_frag_f16(&sVt[32][0], 32, lane, 0);
    v16h bV3 = load_frag_f16(&sVt[48][0], 32, lane, 0);
    O0 = __builtin_amdgcn_wmma_f32_16x16x32_f16(false, aP, false, bV0,
                                                (short)0, O0, false, false);
    O1 = __builtin_amdgcn_wmma_f32_16x16x32_f16(false, aP, false, bV1,
                                                (short)0, O1, false, false);
    O2 = __builtin_amdgcn_wmma_f32_16x16x32_f16(false, aP, false, bV2,
                                                (short)0, O2, false, false);
    O3 = __builtin_amdgcn_wmma_f32_16x16x32_f16(false, aP, false, bV3,
                                                (short)0, O3, false, false);
  }

#pragma unroll
  for (int r = 0; r < 8; ++r) {
    const size_t row = (size_t)(b * TT + qbase + r + hh * 8);
    const float invl = 1.0f / lsum[r];
    float* orow = o + row * (NH * HD) + h * HD;
    orow[0 + nn]  = O0[r] * invl;
    orow[16 + nn] = O1[r] * invl;
    orow[32 + nn] = O2[r] * invl;
    orow[48 + nn] = O3[r] * invl;
  }
}

extern "C" void kernel_launch(void* const* d_in, const int* in_sizes, int n_in,
                              void* d_out, int out_size, void* d_ws,
                              size_t ws_size, hipStream_t stream) {
  (void)in_sizes; (void)n_in; (void)out_size; (void)ws_size;

  const int* idx      = (const int*)d_in[0];
  const int* targets  = (const int*)d_in[1];
  const float* embed  = (const float*)d_in[2];
  const float* pose   = (const float*)d_in[3];
  const float* ln1_w  = (const float*)d_in[4];
  const float* wq     = (const float*)d_in[5];
  const float* wk     = (const float*)d_in[6];
  const float* wv     = (const float*)d_in[7];
  const float* wo     = (const float*)d_in[8];
  const float* ln2_w  = (const float*)d_in[9];
  const float* wg     = (const float*)d_in[10];
  const float* wu     = (const float*)d_in[11];
  const float* wd     = (const float*)d_in[12];
  const float* lnf_w  = (const float*)d_in[13];
  const float* w_lm   = (const float*)d_in[14];

  float* logits = (float*)d_out;
  float* lossP  = logits + (size_t)MM * VV;

  char* ws = (char*)d_ws;
  size_t off = 0;
  auto alloc = [&](size_t bytes) -> char* {
    char* p = ws + off;
    off = (off + bytes + 255) & ~(size_t)255;
    return p;
  };
  float* x   = (float*)alloc((size_t)MM * DD * 4);
  float* hb  = (float*)alloc((size_t)MM * DD * 4);
  float* qb  = (float*)alloc((size_t)MM * DD * 4);
  float* kb  = (float*)alloc((size_t)MM * (NKV * HD) * 4);
  float* vb  = (float*)alloc((size_t)MM * (NKV * HD) * 4);
  float* ao  = (float*)alloc((size_t)MM * DD * 4);
  float* gb  = (float*)alloc((size_t)MM * FFN * 4);
  float* ub  = (float*)alloc((size_t)MM * FFN * 4);
  signed char* xq8 = (signed char*)alloc((size_t)MM * FFN);
  float* xsc = (float*)alloc((size_t)MM * 4);
  float* partials = (float*)alloc(256 * 4);
  float* wsc = (float*)alloc(64 * 4);

  signed char *wq8[LL], *wk8[LL], *wv8[LL], *wo8[LL], *wg8[LL], *wu8[LL], *wd8[LL];
  for (int l = 0; l < LL; ++l) {
    wq8[l] = (signed char*)alloc((size_t)DD * DD);
    wk8[l] = (signed char*)alloc((size_t)(NKV * HD) * DD);
    wv8[l] = (signed char*)alloc((size_t)(NKV * HD) * DD);
    wo8[l] = (signed char*)alloc((size_t)DD * DD);
    wg8[l] = (signed char*)alloc((size_t)FFN * DD);
    wu8[l] = (signed char*)alloc((size_t)FFN * DD);
    wd8[l] = (signed char*)alloc((size_t)DD * FFN);
  }
  signed char* lm8 = (signed char*)alloc((size_t)VV * DD);

  auto quantw = [&](const float* w, size_t n, signed char* q8, int si) {
    wabs_reduce_kernel<<<256, 256, 0, stream>>>(w, (long long)n, partials);
    wquant_kernel<<<256, 256, 0, stream>>>(w, (long long)n, partials, q8, wsc + si);
  };
  for (int l = 0; l < LL; ++l) {
    quantw(wq + (size_t)l * DD * DD, (size_t)DD * DD, wq8[l], l * 7 + 0);
    quantw(wk + (size_t)l * (NKV * HD) * DD, (size_t)(NKV * HD) * DD, wk8[l], l * 7 + 1);
    quantw(wv + (size_t)l * (NKV * HD) * DD, (size_t)(NKV * HD) * DD, wv8[l], l * 7 + 2);
    quantw(wo + (size_t)l * DD * DD, (size_t)DD * DD, wo8[l], l * 7 + 3);
    quantw(wg + (size_t)l * FFN * DD, (size_t)FFN * DD, wg8[l], l * 7 + 4);
    quantw(wu + (size_t)l * FFN * DD, (size_t)FFN * DD, wu8[l], l * 7 + 5);
    quantw(wd + (size_t)l * DD * FFN, (size_t)DD * FFN, wd8[l], l * 7 + 6);
  }
  quantw(w_lm, (size_t)VV * DD, lm8, 42);

  auto gemm = [&](const signed char* Aq, const float* asc, const signed char* Bq,
                  const float* wscp, const float* res, float* Cout, int Mm,
                  int Nn, int Kk, int useRes) {
    dim3 g(Nn / 64, Mm / 64);
    gemm_i8_kernel<<<g, 256, 0, stream>>>(Aq, asc, Bq, wscp, res, Cout, Mm, Nn,
                                          Kk, useRes);
  };

  embed_kernel<<<MM, 256, 0, stream>>>(idx, embed, pose, x);

  for (int l = 0; l < LL; ++l) {
    rmsnorm_kernel<<<MM, 256, 0, stream>>>(x, ln1_w + (size_t)l * DD, hb, DD);
    act_quant_kernel<<<MM, 256, 0, stream>>>(hb, xq8, xsc, DD);
    gemm(xq8, xsc, wq8[l], wsc + l * 7 + 0, nullptr, qb, MM, DD, DD, 0);
    gemm(xq8, xsc, wk8[l], wsc + l * 7 + 1, nullptr, kb, MM, NKV * HD, DD, 0);
    gemm(xq8, xsc, wv8[l], wsc + l * 7 + 2, nullptr, vb, MM, NKV * HD, DD, 0);
    {
      dim3 g(TT / 16, NH, BB);
      flash_attn_kernel<<<g, 32, 0, stream>>>(qb, kb, vb, ao);
    }
    act_quant_kernel<<<MM, 256, 0, stream>>>(ao, xq8, xsc, DD);
    gemm(xq8, xsc, wo8[l], wsc + l * 7 + 3, x, x, MM, DD, DD, 1);

    rmsnorm_kernel<<<MM, 256, 0, stream>>>(x, ln2_w + (size_t)l * DD, hb, DD);
    act_quant_kernel<<<MM, 256, 0, stream>>>(hb, xq8, xsc, DD);
    gemm(xq8, xsc, wg8[l], wsc + l * 7 + 4, nullptr, gb, MM, FFN, DD, 0);
    gemm(xq8, xsc, wu8[l], wsc + l * 7 + 5, nullptr, ub, MM, FFN, DD, 0);
    silu_mul_kernel<<<2048, 256, 0, stream>>>(gb, ub, (long long)MM * FFN);
    act_quant_kernel<<<MM, 256, 0, stream>>>(gb, xq8, xsc, FFN);
    gemm(xq8, xsc, wd8[l], wsc + l * 7 + 6, x, x, MM, DD, FFN, 1);
  }

  rmsnorm_kernel<<<MM, 256, 0, stream>>>(x, lnf_w, hb, DD);
  act_quant_kernel<<<MM, 256, 0, stream>>>(hb, xq8, xsc, DD);
  gemm(xq8, xsc, lm8, wsc + 42, nullptr, logits, MM, VV, DD, 0);

  zero_loss_kernel<<<1, 32, 0, stream>>>(lossP);
  loss_kernel<<<MM, 256, 0, stream>>>(logits, targets, lossP, 1.0f / (float)MM);
}
